// InterChannelGCN_46961172415330
// MI455X (gfx1250) — matrix operand order repaired and checked by reading, not compiled, not run on hardware
//
#include <hip/hip_runtime.h>

typedef __attribute__((ext_vector_type(16))) __bf16 v16bf;
typedef __attribute__((ext_vector_type(8)))  float  v8f;
typedef __attribute__((ext_vector_type(4)))  float  f32x4;
typedef __attribute__((ext_vector_type(4)))  unsigned u32x4;
typedef __attribute__((ext_vector_type(2)))  unsigned u32x2;
typedef __attribute__((ext_vector_type(8)))  int v8i_t;
typedef __attribute__((ext_vector_type(4)))  int v4i_t;

constexpr int N = 16384, M = 4096, D = 256, O = 256;
constexpr int BM  = 64;   // rows of N per block in fused kernel
constexpr int KC  = 32;   // WMMA bf16 K
constexpr int NCH = M / KC;
constexpr int P_LD = 40;  // P tile LDS row stride (ushorts): conflict-free, 16B-mult
constexpr int H_LD = 40;  // HcT LDS row stride (ushorts): 32 data + 4-DWORD pad
constexpr int A_LD = 264; // attn LDS row stride (ushorts)
constexpr int P_BYTES = BM * P_LD * 2;   // 5120
constexpr int H_BYTES = D * H_LD * 2;    // 20480
constexpr int P_ELEMS = P_BYTES / 2;     // 2560
constexpr int H_ELEMS = H_BYTES / 2;     // 10240
constexpr int LDS_BYTES = 2 * (P_BYTES + H_BYTES); // 51200 (>= 64*A_LD*2 = 33792)

#if __has_builtin(__builtin_amdgcn_tensor_load_to_lds)
#define USE_TDM 1
#else
#define USE_TDM 0
#endif

struct Frag32 { u32x4 lo, hi; };

static __device__ __forceinline__ unsigned short f2bf(float f) {
  unsigned u = __builtin_bit_cast(unsigned, f);
  u += 0x7fffu + ((u >> 16) & 1u);          // round-to-nearest-even
  return (unsigned short)(u >> 16);
}
static __device__ __forceinline__ unsigned pack2(float a, float b) {
  return (unsigned)f2bf(a) | ((unsigned)f2bf(b) << 16);
}
static __device__ __forceinline__ v16bf frag_ld(const unsigned short* lo,
                                                const unsigned short* hi) {
  Frag32 f;
  f.lo = *(const u32x4*)lo;
  f.hi = *(const u32x4*)hi;
  return __builtin_bit_cast(v16bf, f);
}
static __device__ __forceinline__ v8f vzero8() {
  v8f z;
#pragma unroll
  for (int q = 0; q < 8; ++q) z[q] = 0.0f;
  return z;
}

// ---------------- kernel 1: alpha[N] = Hf.wa[:D] + ba ; sc[M] = Hc.wa[D:] ----
__global__ void k_scores(const float* __restrict__ Hf, const float* __restrict__ Hc,
                         const float* __restrict__ wa, const float* __restrict__ ba,
                         float* __restrict__ alpha, float* __restrict__ sc) {
  __shared__ float red[256];
  const int tid = threadIdx.x, lane = tid & 31;
  const int row = blockIdx.x * 8 + (tid >> 5);
  const bool isF = row < N;
  const int r = isF ? row : row - N;
  const float* X = isF ? (Hf + (size_t)r * D) : (Hc + (size_t)r * D);
  const float* w = isF ? wa : (wa + D);
  f32x4 x0 = *(const f32x4*)(X + lane * 8);
  f32x4 x1 = *(const f32x4*)(X + lane * 8 + 4);
  f32x4 w0 = *(const f32x4*)(w + lane * 8);
  f32x4 w1 = *(const f32x4*)(w + lane * 8 + 4);
  float p = x0.x*w0.x + x0.y*w0.y + x0.z*w0.z + x0.w*w0.w
          + x1.x*w1.x + x1.y*w1.y + x1.z*w1.z + x1.w*w1.w;
  red[tid] = p;
  __syncthreads();
  for (int s = 16; s > 0; s >>= 1) {
    if (lane < s) red[tid] += red[tid + s];
    __syncthreads();
  }
  if (lane == 0) {
    float v = red[tid];
    if (isF) alpha[r] = v + ba[0];
    else     sc[r]    = v;
  }
}

// ---------------- kernel 1b: HcTg[d][j] = bf16(Hc[j][d])  (transposed) -------
__global__ void k_cvt_hct(const float* __restrict__ Hc, unsigned short* __restrict__ HcTg) {
  const int d  = blockIdx.x;           // 0..D-1
  const int j0 = threadIdx.x * 16;
  float v[16];
#pragma unroll
  for (int e = 0; e < 16; ++e) v[e] = Hc[(size_t)(j0 + e) * D + d];
  u32x4 a, b;
  a.x = pack2(v[0], v[1]);  a.y = pack2(v[2], v[3]);
  a.z = pack2(v[4], v[5]);  a.w = pack2(v[6], v[7]);
  b.x = pack2(v[8], v[9]);  b.y = pack2(v[10], v[11]);
  b.z = pack2(v[12], v[13]); b.w = pack2(v[14], v[15]);
  unsigned short* dst = HcTg + (size_t)d * M + j0;
  *(u32x4*)(dst)     = a;
  *(u32x4*)(dst + 8) = b;
}

// ---------------- kernel 1c: WtB = bf16(Wt) --------------------------------
__global__ void k_cvt_wt(const float* __restrict__ Wt, unsigned short* __restrict__ WtB) {
  const int gid = blockIdx.x * 256 + threadIdx.x;
  f32x4 v = *(const f32x4*)(Wt + (size_t)gid * 4);
  u32x2 p;
  p.x = pack2(v.x, v.y);
  p.y = pack2(v.z, v.w);
  *(u32x2*)(WtB + (size_t)gid * 4) = p;
}

// ---------------- kernel 2: per-row softmax stats over masked scores ---------
__global__ void k_stats(const float* __restrict__ adj, const float* __restrict__ alpha,
                        const float* __restrict__ sc, float* __restrict__ rowmax,
                        float* __restrict__ invsum) {
  __shared__ float red[256];
  const int tid = threadIdx.x;
  const int row = blockIdx.x;
  const float* arow = adj + (size_t)row * M;
  const float al = alpha[row];
  float v[16];
  float lm = -3.0e38f;
#pragma unroll
  for (int k = 0; k < 16; ++k) {
    const int j = tid + (k << 8);
    const float a = __builtin_nontemporal_load(arow + j);  // streamed, > L2
    const float m = a * (al + sc[j]);                      // exact: adj is 0.0 or 1.0
    v[k] = m;
    lm = fmaxf(lm, m);
  }
  red[tid] = lm; __syncthreads();
  for (int s = 128; s > 0; s >>= 1) {
    if (tid < s) red[tid] = fmaxf(red[tid], red[tid + s]);
    __syncthreads();
  }
  const float rmax = red[0];
  __syncthreads();
  float ls = 0.0f;
#pragma unroll
  for (int k = 0; k < 16; ++k) ls += __expf(v[k] - rmax);
  red[tid] = ls; __syncthreads();
  for (int s = 128; s > 0; s >>= 1) {
    if (tid < s) red[tid] += red[tid + s];
    __syncthreads();
  }
  if (tid == 0) { rowmax[row] = rmax; invsum[row] = 1.0f / red[0]; }
}

// ---------------- helpers for fused kernel ----------------------------------
static __device__ __forceinline__ void stage_P(unsigned short* Pbuf, const float* adjRow,
                                               const float* scv, int jc, int prow, int pcol,
                                               float al, float rm, float is) {
  const float* ap = adjRow + jc + pcol;
  f32x4 a0 = __builtin_nontemporal_load((const f32x4*)(ap));
  f32x4 a1 = __builtin_nontemporal_load((const f32x4*)(ap + 4));
  f32x4 s0 = *(const f32x4*)(scv + jc + pcol);
  f32x4 s1 = *(const f32x4*)(scv + jc + pcol + 4);
  float e0 = __expf(a0.x * (al + s0.x) - rm) * is;
  float e1 = __expf(a0.y * (al + s0.y) - rm) * is;
  float e2 = __expf(a0.z * (al + s0.z) - rm) * is;
  float e3 = __expf(a0.w * (al + s0.w) - rm) * is;
  float e4 = __expf(a1.x * (al + s1.x) - rm) * is;
  float e5 = __expf(a1.y * (al + s1.y) - rm) * is;
  float e6 = __expf(a1.z * (al + s1.z) - rm) * is;
  float e7 = __expf(a1.w * (al + s1.w) - rm) * is;
  u32x4 pk;
  pk.x = pack2(e0, e1); pk.y = pack2(e2, e3);
  pk.z = pack2(e4, e5); pk.w = pack2(e6, e7);
  *(u32x4*)(Pbuf + prow * P_LD + pcol) = pk;
}

#if USE_TDM
// Issue a TDM load of a [256 rows x 64B] tile (row stride M*2 bytes) into LDS,
// with 16B pad after every 64B row -> LDS row stride H_LD (=40 ushorts).
static __device__ __forceinline__ void tdm_issue(unsigned lds_addr, const void* gsrc) {
  unsigned long long ga = (unsigned long long)(size_t)gsrc;
  u32x4 g0;
  g0.x = 1u;                                   // count=1 (valid user descriptor)
  g0.y = lds_addr;                             // D# bits 63:32 : LDS byte address
  g0.z = (unsigned)ga;                         // global addr low
  g0.w = (unsigned)((ga >> 32) & 0x1ffffffull) // global addr bits 56:32
       | (2u << 30);                           // type = 2 ("image")
  v8i_t g1;
  g1[0] = (int)((1u << 16)      // data_size = 2 bytes
              | (1u << 20)      // pad_enable
              | (3u << 22)      // pad_interval: every 16 DWORDs (64B)
              | (3u << 25));    // pad_amount: 4 DWORDs (16B)
  g1[1] = (int)(((unsigned)M & 0xffffu) << 16);                       // tensor_dim0 lo16
  g1[2] = (int)((((unsigned)M >> 16) & 0xffffu)
              | (((unsigned)D & 0xffffu) << 16));                     // dim0 hi16 | dim1 lo16
  g1[3] = (int)((((unsigned)D >> 16) & 0xffffu) | ((unsigned)KC << 16)); // dim1 hi | tile_dim0=32
  g1[4] = (int)(unsigned)D;     // tile_dim1 = 256 rows ; tile_dim2 = 0
  g1[5] = (int)(unsigned)M;     // tensor_dim0_stride (elements), low 32
  g1[6] = 0;
  g1[7] = 0;
  v4i_t z4; z4[0] = 0; z4[1] = 0; z4[2] = 0; z4[3] = 0;
  v8i_t z8;
#pragma unroll
  for (int q = 0; q < 8; ++q) z8[q] = 0;
  // 6-arg toolchain form: (g0, g1, g2, g3, g4, cpol)
  __builtin_amdgcn_tensor_load_to_lds(g0, g1, z4, z4, z8, 0);
}
#else
static __device__ __forceinline__ void stage_Hc_manual(unsigned short* Hbuf,
                                                       const unsigned short* HcTg,
                                                       int jc, int tid) {
  const unsigned short* src = HcTg + (size_t)tid * M + jc;
  u32x4 a = *(const u32x4*)(src);
  u32x4 b = *(const u32x4*)(src + 8);
  u32x4 c = *(const u32x4*)(src + 16);
  u32x4 d = *(const u32x4*)(src + 24);
  unsigned short* dst = Hbuf + tid * H_LD;
  *(u32x4*)(dst)      = a;
  *(u32x4*)(dst + 8)  = b;
  *(u32x4*)(dst + 16) = c;
  *(u32x4*)(dst + 24) = d;
}
#endif

// ---------------- kernel 3: fused softmax-weights GEMM + epilogue GEMM -------
__global__ __launch_bounds__(256)
void k_fused(const float* __restrict__ adj, const unsigned short* __restrict__ HcTg,
             const float* __restrict__ Hf, const unsigned short* __restrict__ WtB,
             const float* __restrict__ bt, const float* __restrict__ alpha,
             const float* __restrict__ sc, const float* __restrict__ rowmax,
             const float* __restrict__ invsum, float* __restrict__ out) {
  __shared__ __align__(16) unsigned char smem[LDS_BYTES];
  unsigned short* const Pbase = (unsigned short*)smem;
  unsigned short* const Hbase = (unsigned short*)(smem + 2 * P_BYTES);
  unsigned short* const Alds  = (unsigned short*)smem;      // epilogue reuse

  const int tid  = threadIdx.x;
  const int lane = tid & 31;
  const int w    = tid >> 5;
  const int r0   = (w & 3) * 16;
  const int cg   = w >> 2;
  const int i0   = blockIdx.x * BM;

  // staging roles
  const int prow = tid >> 2;
  const int pcol = (tid & 3) * 8;
  const float* adjRow = adj + (size_t)(i0 + prow) * M;
  const float p_al = alpha[i0 + prow];
  const float p_rm = rowmax[i0 + prow];
  const float p_is = invsum[i0 + prow];

  // per-lane fragment geometry (16-bit WMMA layouts)
  const int arow = r0 + (lane & 15);
  const int kb   = (lane < 16) ? 0 : 8;
  const int koff = (lane < 16) ? 0 : 16;
  const int bl   = lane & 15;
  const int hi8  = (lane >> 4) * 8;

  v8f acc[8];
#pragma unroll
  for (int t = 0; t < 8; ++t) acc[t] = vzero8();

  // ---- prologue: start pipeline for chunk 0 ----
#if USE_TDM
  if (tid < 32) tdm_issue((unsigned)(size_t)Hbase, HcTg);
#else
  stage_Hc_manual(Hbase, HcTg, 0, tid);
#endif
  stage_P(Pbase, adjRow, sc, 0, prow, pcol, p_al, p_rm, p_is);

  for (int i = 0; i < NCH; ++i) {
    const int cur = i & 1, nxt = cur ^ 1;
    unsigned short* const Pcur = Pbase + cur * P_ELEMS;
    unsigned short* const Hcur = Hbase + cur * H_ELEMS;
    unsigned short* const Pnxt = Pbase + nxt * P_ELEMS;
    unsigned short* const Hnxt = Hbase + nxt * H_ELEMS;
#if USE_TDM
    if (tid < 32) __builtin_amdgcn_s_wait_tensorcnt(0);
#endif
    __syncthreads();   // chunk i staged for all waves; prev reads of [nxt] done
    if (i + 1 < NCH) {
      const int jn = (i + 1) * KC;
#if USE_TDM
      if (tid < 32) tdm_issue((unsigned)(size_t)Hnxt, HcTg + jn);
#else
      stage_Hc_manual(Hnxt, HcTg, jn, tid);
#endif
      stage_P(Pnxt, adjRow, sc, jn, prow, pcol, p_al, p_rm, p_is);
    }
    // ---- WMMA: acc[16x128 per wave] += P[16x32] @ Hc[32x128] ----
    // even/odd two-register B pipeline (no copies -> no coalescing back to
    // one register pair; loads for tile t+2 issue before wmma consumes tile t+1)
    v16bf afrag = frag_ld(Pcur + arow * P_LD + koff,
                          Pcur + arow * P_LD + koff + 8);
    const unsigned short* bbase = Hcur + (cg * 128 + bl) * H_LD + koff;
    v16bf b0 = frag_ld(bbase, bbase + 8);
    v16bf b1 = frag_ld(bbase + 16 * H_LD, bbase + 16 * H_LD + 8);
#pragma unroll
    for (int t = 0; t < 8; t += 2) {
      acc[t] = __builtin_amdgcn_wmma_f32_16x16x32_bf16(
          false, afrag, false, b0, (short)0, acc[t], false, false);
      if (t + 2 < 8) {
        const unsigned short* bp = bbase + (t + 2) * 16 * H_LD;
        b0 = frag_ld(bp, bp + 8);
      }
      acc[t + 1] = __builtin_amdgcn_wmma_f32_16x16x32_bf16(
          false, afrag, false, b1, (short)0, acc[t + 1], false, false);
      if (t + 3 < 8) {
        const unsigned short* bp = bbase + (t + 3) * 16 * H_LD;
        b1 = frag_ld(bp, bp + 8);
      }
    }
  }

  // ---- epilogue part 1: attn = acc + H_fine -> LDS (bf16) ----
  __syncthreads();
#pragma unroll
  for (int t = 0; t < 8; ++t) {
    const int col = cg * 128 + t * 16 + bl;
#pragma unroll
    for (int v = 0; v < 8; ++v) {
      const int row = r0 + hi8 + v;
      const float val = acc[t][v] + Hf[(size_t)(i0 + row) * D + col];
      Alds[row * A_LD + col] = f2bf(val);
    }
  }
  __syncthreads();

  // ---- epilogue part 2: out = relu(attn @ Wt^T + bt) ----
  v8f oacc[8];
#pragma unroll
  for (int t = 0; t < 8; ++t) oacc[t] = vzero8();
#pragma unroll 1
  for (int kc = 0; kc < D; kc += KC) {
    v16bf a2 = frag_ld(Alds + arow * A_LD + kc + koff,
                       Alds + arow * A_LD + kc + koff + 8);
#pragma unroll
    for (int t = 0; t < 8; ++t) {
      const int ocol = cg * 128 + t * 16 + bl;
      const unsigned short* wp = WtB + (size_t)ocol * D + kc + koff;
      v16bf b2 = frag_ld(wp, wp + 8);
      oacc[t] = __builtin_amdgcn_wmma_f32_16x16x32_bf16(
          false, a2, false, b2, (short)0, oacc[t], false, false);
    }
  }
#pragma unroll
  for (int t = 0; t < 8; ++t) {
    const int ocol = cg * 128 + t * 16 + bl;
    const float bias = bt[ocol];
#pragma unroll
    for (int v = 0; v < 8; ++v) {
      const int row = i0 + r0 + hi8 + v;
      __builtin_nontemporal_store(fmaxf(oacc[t][v] + bias, 0.0f),
                                  out + (size_t)row * O + ocol);
    }
  }
}

extern "C" void kernel_launch(void* const* d_in, const int* in_sizes, int n_in,
                              void* d_out, int out_size, void* d_ws, size_t ws_size,
                              hipStream_t stream) {
  (void)in_sizes; (void)n_in; (void)out_size; (void)ws_size;
  const float* Hf  = (const float*)d_in[0];
  const float* Hc  = (const float*)d_in[1];
  const float* adj = (const float*)d_in[2];
  const float* wa  = (const float*)d_in[3];
  const float* ba  = (const float*)d_in[4];
  const float* Wt  = (const float*)d_in[5];
  const float* bt  = (const float*)d_in[6];
  float* out = (float*)d_out;
  float* ws  = (float*)d_ws;

  float* alpha  = ws;               // [N]
  float* sc     = ws + N;           // [M]
  float* rowmax = ws + N + M;       // [N]
  float* invsum = ws + N + M + N;   // [N]
  unsigned short* HcTg = (unsigned short*)(ws + 3 * N + M);  // [D][M] bf16
  unsigned short* WtB  = HcTg + (size_t)D * M;               // [O][D] bf16

  k_scores <<<dim3((N + M) / 8), dim3(256), 0, stream>>>(Hf, Hc, wa, ba, alpha, sc);
  k_cvt_hct<<<dim3(D),           dim3(256), 0, stream>>>(Hc, HcTg);
  k_cvt_wt <<<dim3(O * D / 1024), dim3(256), 0, stream>>>(Wt, WtB);
  k_stats  <<<dim3(N),           dim3(256), 0, stream>>>(adj, alpha, sc, rowmax, invsum);
  k_fused  <<<dim3(N / BM),      dim3(256), 0, stream>>>(adj, HcTg, Hf, WtB, bt,
                                                         alpha, sc, rowmax, invsum, out);
}
